// GAT_22247930593786
// MI455X (gfx1250) — matrix operand order repaired and checked by reading, not compiled, not run on hardware
//
#include <hip/hip_runtime.h>
#include <hip/hip_bf16.h>

// ---------------------------------------------------------------------------
// GAT link predictor for gfx1250 (MI455X), wave32 / WMMA bf16 / TDM pipeline.
//   pack W1,W2 -> bf16 B-fragments | gemm1 (TDM stage + WMMA) | alpha1 |
//   softmax1 (3 edge passes, L2 atomics) | scatter1 (float4) | bias+relu |
//   gemm2 | alpha2 | softmax2 | scatter2 (head-mean folded) | bias | decode
// All intermediates (~150MB) stay L2-resident (192MB L2).
// ---------------------------------------------------------------------------

typedef __attribute__((ext_vector_type(16))) __bf16 v16bf;
typedef __attribute__((ext_vector_type(8)))  float  v8f;

#if defined(__HIP_DEVICE_COMPILE__) && defined(__gfx1250__) && \
    __has_builtin(__builtin_amdgcn_tensor_load_to_lds) &&      \
    __has_builtin(__builtin_amdgcn_s_wait_tensorcnt)
#define GAT_TDM 1
typedef __attribute__((ext_vector_type(4))) unsigned v4u;
typedef __attribute__((ext_vector_type(8))) int v8i;
typedef __attribute__((ext_vector_type(4))) int v4i;
#else
#define GAT_TDM 0
#endif

#define LRELU 0.2f
#define EPSV  1e-16f

union FragU { v16bf v; unsigned u[8]; };

__device__ __forceinline__ v8f wmma_bf16(const FragU& a, const FragU& b, v8f c) {
  return __builtin_amdgcn_wmma_f32_16x16x32_bf16(false, a.v, false, b.v,
                                                 (short)0, c, false, false);
}

__device__ __forceinline__ unsigned pk_bf16x2(float a, float b) {
  unsigned ua = __float_as_uint(a), ub = __float_as_uint(b);
  ua += 0x7FFFu + ((ua >> 16) & 1u);          // round-to-nearest-even
  ub += 0x7FFFu + ((ub >> 16) & 1u);
  return (ua >> 16) | (ub & 0xFFFF0000u);
}

__device__ __forceinline__ void atomicMaxF(float* addr, float val) {
  if (val >= 0.0f) atomicMax((int*)addr, __float_as_int(val));
  else             atomicMin((unsigned*)addr, __float_as_uint(val));
}

// src/dst for edge e; edges >= E are the appended self-loops
__device__ __forceinline__ void edge_sd(int e, const int* __restrict__ ei,
                                        int E, int& s, int& d) {
  if (e < E) { s = ei[e]; d = ei[E + e]; } else { s = d = e - E; }
}

__global__ void k_fill(float* __restrict__ p, float v, long n) {
  long i = (long)blockIdx.x * blockDim.x + threadIdx.x;
  if (i < n) p[i] = v;
}

// ---- pack W [K,NC] f32 row-major -> bf16 B fragments [NT][KT][lane][v] -----
template <int K, int NC>
__global__ void k_pack_b(const float* __restrict__ W, unsigned* __restrict__ Bp) {
  constexpr int KT = K / 32;
  constexpr int TOT = (NC / 16) * KT * 256;
  int idx = blockIdx.x * blockDim.x + threadIdx.x;
  if (idx >= TOT) return;
  int v = idx & 7, l = (idx >> 3) & 31, t = idx >> 8;
  int kt = t % KT, nt = t / KT;
  int n = nt * 16 + (l & 15);
  int k = kt * 32 + ((l >> 4) << 3) + 2 * v + ((v >> 2) << 3);   // lanes>=16 hold K=8..15,24..31
  Bp[idx] = pk_bf16x2(W[(size_t)k * NC + n], W[(size_t)(k + 1) * NC + n]);
}

// ---- C[M,NC] = A[M,K] @ B ; 64-row blocks, 8 waves, B-frag reuse x2 --------
template <int K, int NC>
__global__ __launch_bounds__(256) void k_wmma_gemm(const float* __restrict__ A,
                                                   const unsigned* __restrict__ Bp,
                                                   float* __restrict__ C, int M) {
  constexpr int KT = K / 32;
  constexpr int NT = NC / 16;
  constexpr int FRAG_W = KT * 256;             // u32 per 16-row A-fragment set
  __shared__ unsigned lA[4 * FRAG_W];          // 4 M-tiles of A fragments
#if GAT_TDM
  __shared__ float rawA[64 * K];               // raw f32 tile staged by TDM
#endif
  const int row0 = blockIdx.x * 64;
  const int tid  = threadIdx.x;
  const int wave = tid >> 5;
  const int lane = tid & 31;

#if GAT_TDM
  if (wave == 0) {
    // Tensor DMA: 2D tensor [M rows x K], tile [64 x K]; OOB rows zero-fill.
    unsigned lds_off = (unsigned)(unsigned long long)(uintptr_t)rawA;
    unsigned long long ga =
        (unsigned long long)(uintptr_t)A + (unsigned long long)row0 * K * 4ull;
    v4u g0;
    g0[0] = 1u;                                         // count=1, user mode
    g0[1] = lds_off;                                    // lds_addr
    g0[2] = (unsigned)(ga & 0xFFFFFFFFull);             // global_addr[31:0]
    g0[3] = (unsigned)((ga >> 32) & 0x01FFFFFFull) | (2u << 30);  // type=2
    const unsigned td0 = (unsigned)K, td1 = (unsigned)M;
    const unsigned long long st0 = (unsigned long long)K;
    v8i g1;
    g1[0] = (int)(2u << 16);                            // data_size=4B, mask=0
    g1[1] = (int)((td0 & 0xFFFFu) << 16);               // tensor_dim0[15:0]
    g1[2] = (int)((td0 >> 16) | ((td1 & 0xFFFFu) << 16));
    g1[3] = (int)((td1 >> 16) | ((unsigned)K << 16));   // tile_dim0 = K
    g1[4] = (int)64;                                    // tile_dim1=64, dim2=0
    g1[5] = (int)(st0 & 0xFFFFFFFFull);                 // dim0 stride = K
    g1[6] = (int)((st0 >> 32) & 0xFFFFull);             // dim1 stride = 0
    g1[7] = 0;
    v4i gz = {0, 0, 0, 0};
#if __clang_major__ >= 23
    v8i gz8 = {0, 0, 0, 0, 0, 0, 0, 0};
    __builtin_amdgcn_tensor_load_to_lds(g0, g1, gz, gz, gz8, 0);
#else
    __builtin_amdgcn_tensor_load_to_lds(g0, g1, gz, gz, 0);
#endif
    __builtin_amdgcn_s_wait_tensorcnt(0);
  }
  __syncthreads();
  // convert raw f32 tile (LDS) -> bf16 A fragments (LDS)
  for (int idx = tid; idx < 4 * FRAG_W; idx += 256) {
    int v = idx & 7, l = (idx >> 3) & 31, rest = idx >> 8;
    int kt = rest % KT, mt = rest / KT;
    int m  = mt * 16 + (l & 15);
    int kb = (v < 4) ? (2 * v) : (16 + 2 * (v - 4));
    if (l >= 16) kb += 8;
    const float* ap = rawA + (size_t)m * K + kt * 32 + kb;
    lA[idx] = pk_bf16x2(ap[0], ap[1]);
  }
#else
  for (int idx = tid; idx < 4 * FRAG_W; idx += 256) {
    int v = idx & 7, l = (idx >> 3) & 31, rest = idx >> 8;
    int kt = rest % KT, mt = rest / KT;
    int m  = mt * 16 + (l & 15);
    int kb = (v < 4) ? (2 * v) : (16 + 2 * (v - 4));
    if (l >= 16) kb += 8;
    int row = row0 + m; if (row >= M) row = M - 1;
    const float* ap = A + (size_t)row * K + kt * 32 + kb;
    lA[idx] = pk_bf16x2(ap[0], ap[1]);
  }
  if (tid == 0 && row0 + 64 < M)               // global_prefetch_b8 next tile
    __builtin_prefetch(A + (size_t)(row0 + 64) * K, 0, 1);
#endif
  __syncthreads();

  const bool full = (row0 + 64 <= M);
  // work unit = (N-tile, pair of M-tiles): B fragment reused for 2 WMMAs.
  for (int u = wave; u < NT * 2; u += 8) {     // uniform per wave: EXEC all-1s
    const int nt = u >> 1, mg = u & 1;
    v8f acc0 = {}, acc1 = {};
#pragma unroll
    for (int kt = 0; kt < KT; ++kt) {
      FragU a0, a1, b;
      const unsigned* gb  = &Bp[((size_t)nt * KT + kt) * 256 + lane * 8];
      const unsigned* la0 = &lA[(2 * mg) * FRAG_W + kt * 256 + lane * 8];
      const unsigned* la1 = la0 + FRAG_W;
#pragma unroll
      for (int i = 0; i < 8; ++i) { b.u[i] = gb[i]; a0.u[i] = la0[i]; a1.u[i] = la1[i]; }
      acc0 = wmma_bf16(a0, b, acc0);
      acc1 = wmma_bf16(a1, b, acc1);
    }
    const int n   = nt * 16 + (lane & 15);
    const int mb0 = row0 + (2 * mg) * 16 + ((lane >> 4) << 3);
    float* c0 = C + (size_t)mb0 * NC + n;      // VGPR r -> M = r + 8*(lane>=16)
    float* c1 = c0 + (size_t)16 * NC;
    if (full) {
#pragma unroll
      for (int r = 0; r < 8; ++r) {            // immediate-offset stores
        c0[(size_t)r * NC] = acc0[r];
        c1[(size_t)r * NC] = acc1[r];
      }
    } else {
#pragma unroll
      for (int r = 0; r < 8; ++r) {
        if (mb0 + r      < M) c0[(size_t)r * NC] = acc0[r];
        if (mb0 + 16 + r < M) c1[(size_t)r * NC] = acc1[r];
      }
    }
  }
}

// ---- per-node attention logits --------------------------------------------
__global__ void k_alpha1(const float* __restrict__ h, const float* __restrict__ asr,
                         const float* __restrict__ adt, float* __restrict__ os,
                         float* __restrict__ od, int N) {
  int n = blockIdx.x * blockDim.x + threadIdx.x;
  if (n >= N) return;
  const float4* hp = (const float4*)(h + (size_t)n * 64);
  float s = 0.f, d = 0.f;
#pragma unroll
  for (int q = 0; q < 16; ++q) {
    float4 hv = hp[q];
    s += hv.x * asr[4 * q] + hv.y * asr[4 * q + 1] + hv.z * asr[4 * q + 2] + hv.w * asr[4 * q + 3];
    d += hv.x * adt[4 * q] + hv.y * adt[4 * q + 1] + hv.z * adt[4 * q + 2] + hv.w * adt[4 * q + 3];
  }
  os[n] = s; od[n] = d;
}

__global__ void k_alpha2(const float* __restrict__ h, const float* __restrict__ asr,
                         const float* __restrict__ adt, float* __restrict__ os,
                         float* __restrict__ od, int N) {
  int t = blockIdx.x * blockDim.x + threadIdx.x;
  if (t >= N * 8) return;
  int n = t >> 3, hh = t & 7;
  const float4* hp = (const float4*)(h + (size_t)n * 512 + hh * 64);
  const float4* sp = (const float4*)(asr + hh * 64);
  const float4* dp = (const float4*)(adt + hh * 64);
  float s = 0.f, d = 0.f;
#pragma unroll
  for (int q = 0; q < 16; ++q) {
    float4 hv = hp[q], sv = sp[q], dv = dp[q];
    s += hv.x * sv.x + hv.y * sv.y + hv.z * sv.z + hv.w * sv.w;
    d += hv.x * dv.x + hv.y * dv.y + hv.z * dv.z + hv.w * dv.w;
  }
  os[t] = s; od[t] = d;
}

// ---- segment softmax, H heads (H=1 or 8) ----------------------------------
template <int H>
__global__ void k_edge_max(const int* __restrict__ ei, int E, int E2,
                           const float* __restrict__ as, const float* __restrict__ ad,
                           float* __restrict__ ew, float* __restrict__ emax) {
  int t = blockIdx.x * blockDim.x + threadIdx.x;
  if (t >= E2 * H) return;
  int e = t / H, h = t % H, s, d;
  edge_sd(e, ei, E, s, d);
  float v = as[s * H + h] + ad[d * H + h];
  v = (v > 0.f) ? v : LRELU * v;
  ew[t] = v;
  atomicMaxF(&emax[d * H + h], v);
}

template <int H>
__global__ void k_edge_exp(const int* __restrict__ ei, int E, int E2,
                           float* __restrict__ ew, const float* __restrict__ emax,
                           float* __restrict__ den) {
  int t = blockIdx.x * blockDim.x + threadIdx.x;
  if (t >= E2 * H) return;
  int e = t / H, h = t % H, s, d;
  edge_sd(e, ei, E, s, d);
  float v = __expf(ew[t] - emax[d * H + h]);
  ew[t] = v;
  atomicAdd(&den[d * H + h], v);
}

template <int H>
__global__ void k_edge_alpha(const int* __restrict__ ei, int E, int E2,
                             float* __restrict__ ew, const float* __restrict__ den) {
  int t = blockIdx.x * blockDim.x + threadIdx.x;
  if (t >= E2 * H) return;
  int e = t / H, h = t % H, s, d;
  edge_sd(e, ei, E, s, d);
  ew[t] = ew[t] / (den[d * H + h] + EPSV);
}

// ---- message scatter (float4 gathers, scalar f32 atomics to L2) ------------
__global__ void k_scatter1(const int* __restrict__ ei, int E, int E2,
                           const float* __restrict__ h1, const float* __restrict__ ew,
                           float* __restrict__ z) {
  long t = (long)blockIdx.x * blockDim.x + threadIdx.x;
  if (t >= (long)E2 * 16) return;
  int e = (int)(t >> 4), c4 = (int)(t & 15) << 2, s, d;
  edge_sd(e, ei, E, s, d);
  float a = ew[e];
  float4 hv = *(const float4*)(h1 + (size_t)s * 64 + c4);
  float* zp = z + (size_t)d * 64 + c4;
  atomicAdd(zp + 0, hv.x * a);
  atomicAdd(zp + 1, hv.y * a);
  atomicAdd(zp + 2, hv.z * a);
  atomicAdd(zp + 3, hv.w * a);
}

// heads=8, concat=False: head-mean folded into the scatter.
__global__ void k_scatter2(const int* __restrict__ ei, int E, int E2,
                           const float* __restrict__ h2, const float* __restrict__ ew,
                           float* __restrict__ z) {
  long t = (long)blockIdx.x * blockDim.x + threadIdx.x;
  if (t >= (long)E2 * 16) return;
  int e = (int)(t >> 4), c4 = (int)(t & 15) << 2, s, d;
  edge_sd(e, ei, E, s, d);
  const float* hr = h2 + (size_t)s * 512 + c4;
  const float* ar = ew + (size_t)e * 8;
  float4 acc = {0.f, 0.f, 0.f, 0.f};
#pragma unroll
  for (int h = 0; h < 8; ++h) {
    float a = ar[h];
    float4 hv = *(const float4*)(hr + h * 64);
    acc.x += a * hv.x; acc.y += a * hv.y; acc.z += a * hv.z; acc.w += a * hv.w;
  }
  float* zp = z + (size_t)d * 64 + c4;
  atomicAdd(zp + 0, acc.x * 0.125f);
  atomicAdd(zp + 1, acc.y * 0.125f);
  atomicAdd(zp + 2, acc.z * 0.125f);
  atomicAdd(zp + 3, acc.w * 0.125f);
}

__global__ void k_bias_relu(float* __restrict__ z, const float* __restrict__ b,
                            int N, int do_relu) {
  long t = (long)blockIdx.x * blockDim.x + threadIdx.x;
  if (t >= (long)N * 16) return;
  int c4 = (int)(t & 15) << 2;
  float4 v = *(float4*)(z + t * 4);
  v.x += b[c4]; v.y += b[c4 + 1]; v.z += b[c4 + 2]; v.w += b[c4 + 3];
  if (do_relu) {
    v.x = fmaxf(v.x, 0.f); v.y = fmaxf(v.y, 0.f);
    v.z = fmaxf(v.z, 0.f); v.w = fmaxf(v.w, 0.f);
  }
  *(float4*)(z + t * 4) = v;
}

// ---- link decode: dot(z[i], z[j]) -----------------------------------------
__global__ void k_decode(const float* __restrict__ z, const int* __restrict__ eli,
                         int EL, float* __restrict__ out) {
  int t = blockIdx.x * blockDim.x + threadIdx.x;
  if (t >= EL) return;
  const float4* a = (const float4*)(z + (size_t)eli[t] * 64);
  const float4* b = (const float4*)(z + (size_t)eli[EL + t] * 64);
  float s = 0.f;
#pragma unroll
  for (int q = 0; q < 16; ++q) {
    float4 av = a[q], bv = b[q];
    s += av.x * bv.x + av.y * bv.y + av.z * bv.z + av.w * bv.w;
  }
  out[t] = s;
}

// ---------------------------------------------------------------------------
static inline unsigned cdiv(long n, int b) { return (unsigned)((n + b - 1) / b); }

extern "C" void kernel_launch(void* const* d_in, const int* in_sizes, int n_in,
                              void* d_out, int out_size, void* d_ws, size_t ws_size,
                              hipStream_t stream) {
  (void)n_in; (void)out_size; (void)ws_size;
  const float* x    = (const float*)d_in[0];
  const int*   ei   = (const int*)d_in[1];
  const int*   eli  = (const int*)d_in[2];
  const float* W1   = (const float*)d_in[3];
  const float* as1w = (const float*)d_in[4];
  const float* ad1w = (const float*)d_in[5];
  const float* b1   = (const float*)d_in[6];
  const float* W2   = (const float*)d_in[7];
  const float* as2w = (const float*)d_in[8];
  const float* ad2w = (const float*)d_in[9];
  const float* b2   = (const float*)d_in[10];
  float* out = (float*)d_out;

  const int N  = in_sizes[0] / 128;   // 50000
  const int E  = in_sizes[1] / 2;     // 400000
  const int EL = in_sizes[2] / 2;     // 100000
  const int E2 = E + N;               // with self loops

  // bump-allocate workspace (256B aligned)
  char* p = (char*)d_ws;
  auto alloc = [&](size_t bytes) -> void* {
    void* r = (void*)p; p += (bytes + 255) & ~(size_t)255; return r;
  };
  unsigned* B1p  = (unsigned*)alloc((size_t)(128 * 64 / 2) * 4);
  unsigned* B2p  = (unsigned*)alloc((size_t)(64 * 512 / 2) * 4);
  float* h1    = (float*)alloc((size_t)N * 64 * 4);
  float* z1    = (float*)alloc((size_t)N * 64 * 4);
  float* as1   = (float*)alloc((size_t)N * 4);
  float* ad1   = (float*)alloc((size_t)N * 4);
  float* emax1 = (float*)alloc((size_t)N * 4);
  float* den1  = (float*)alloc((size_t)N * 4);
  float* ew1   = (float*)alloc((size_t)E2 * 4);
  float* h2    = (float*)alloc((size_t)N * 512 * 4);
  float* as2   = (float*)alloc((size_t)N * 8 * 4);
  float* ad2   = (float*)alloc((size_t)N * 8 * 4);
  float* emax2 = (float*)alloc((size_t)N * 8 * 4);
  float* den2  = (float*)alloc((size_t)N * 8 * 4);
  float* ew2   = (float*)alloc((size_t)E2 * 8 * 4);
  float* z2    = (float*)alloc((size_t)N * 64 * 4);

  const float NEG_INF = -__builtin_huge_valf();

  // weight packing (bf16 B-fragment order)
  k_pack_b<128, 64><<<cdiv(128 * 64 / 2, 256), 256, 0, stream>>>(W1, B1p);
  k_pack_b<64, 512><<<cdiv(64 * 512 / 2, 256), 256, 0, stream>>>(W2, B2p);

  // init accumulators
  k_fill<<<cdiv((long)N * 64, 256), 256, 0, stream>>>(z1, 0.f, (long)N * 64);
  k_fill<<<cdiv((long)N * 64, 256), 256, 0, stream>>>(z2, 0.f, (long)N * 64);
  k_fill<<<cdiv(N, 256), 256, 0, stream>>>(emax1, NEG_INF, N);
  k_fill<<<cdiv(N, 256), 256, 0, stream>>>(den1, 0.f, N);
  k_fill<<<cdiv((long)N * 8, 256), 256, 0, stream>>>(emax2, NEG_INF, (long)N * 8);
  k_fill<<<cdiv((long)N * 8, 256), 256, 0, stream>>>(den2, 0.f, (long)N * 8);

  // ---- layer 1 ----
  k_wmma_gemm<128, 64><<<cdiv(N, 64), 256, 0, stream>>>(x, B1p, h1, N);
  k_alpha1<<<cdiv(N, 256), 256, 0, stream>>>(h1, as1w, ad1w, as1, ad1, N);
  k_edge_max<1><<<cdiv(E2, 256), 256, 0, stream>>>(ei, E, E2, as1, ad1, ew1, emax1);
  k_edge_exp<1><<<cdiv(E2, 256), 256, 0, stream>>>(ei, E, E2, ew1, emax1, den1);
  k_edge_alpha<1><<<cdiv(E2, 256), 256, 0, stream>>>(ei, E, E2, ew1, den1);
  k_scatter1<<<cdiv((long)E2 * 16, 256), 256, 0, stream>>>(ei, E, E2, h1, ew1, z1);
  k_bias_relu<<<cdiv((long)N * 16, 256), 256, 0, stream>>>(z1, b1, N, 1);

  // ---- layer 2 ----
  k_wmma_gemm<64, 512><<<cdiv(N, 64), 256, 0, stream>>>(z1, B2p, h2, N);
  k_alpha2<<<cdiv((long)N * 8, 256), 256, 0, stream>>>(h2, as2w, ad2w, as2, ad2, N);
  k_edge_max<8><<<cdiv((long)E2 * 8, 256), 256, 0, stream>>>(ei, E, E2, as2, ad2, ew2, emax2);
  k_edge_exp<8><<<cdiv((long)E2 * 8, 256), 256, 0, stream>>>(ei, E, E2, ew2, emax2, den2);
  k_edge_alpha<8><<<cdiv((long)E2 * 8, 256), 256, 0, stream>>>(ei, E, E2, ew2, den2);
  k_scatter2<<<cdiv((long)E2 * 16, 256), 256, 0, stream>>>(ei, E, E2, h2, ew2, z2);
  k_bias_relu<<<cdiv((long)N * 16, 256), 256, 0, stream>>>(z2, b2, N, 0);

  // ---- decode ----
  k_decode<<<cdiv(EL, 256), 256, 0, stream>>>(z2, eli, EL, out);
}
